// MixtureOfExperts_30485677867979
// MI455X (gfx1250) — compile-verified, matrix operand order checked by reading, not run on hardware
//
#include <hip/hip_runtime.h>
#include <math.h>
#include <stdint.h>

// Problem constants (match reference)
#define N_TOK 8192
#define DDIM  1024
#define NEXP  8
#define ODIM  1024
#define HDIM  256

// LDS row stride for f16 tiles: 32 K-halves + 8 pad -> 80 bytes.
// 16B aligned (80 = 5*16) and bank-conflict-free for the fragment pattern
// (row*20 mod 64 distinct over 16 rows). The TDM descriptor reproduces this
// via pad_interval = 16 DWORDs (64B), pad_amount = 4 DWORDs (16B).
#define LDS_STRIDE 40

typedef __attribute__((ext_vector_type(8)))  _Float16 v8h;
typedef __attribute__((ext_vector_type(16))) _Float16 v16h;
typedef __attribute__((ext_vector_type(8)))  float    v8f;

union HF16 { v16h v; v8h h[2]; };

// ---------------------------------------------------------------------------
// Tensor Data Mover support (gfx1250). Guarded: falls back to synchronous
// per-thread copies when the builtin is unavailable.
// This toolchain (clang-23 / therock) exposes the 6-arg builtin:
//   (uint32x4 g0, int32x8 g1, int32x4 g2, int32x4 g3, int32x8 g4, i32 cpol)
// ---------------------------------------------------------------------------
#if defined(__has_builtin)
#if __has_builtin(__builtin_amdgcn_tensor_load_to_lds) && \
    __has_builtin(__builtin_amdgcn_s_wait_tensorcnt)
#define USE_TDM 1
#endif
#endif
#ifndef USE_TDM
#define USE_TDM 0
#endif

#if USE_TDM
typedef __attribute__((ext_vector_type(4))) unsigned int u32x4;
typedef __attribute__((ext_vector_type(8))) int i32x8;
typedef __attribute__((ext_vector_type(4))) int i32x4;

// Low 32 bits of a flat LDS address are the DS offset (ISA 10.2 aperture map).
__device__ __forceinline__ unsigned lds_off(const void* p) {
  return (unsigned)(unsigned long long)(uintptr_t)p;
}

// Issue one TDM tile load: up-to-3D tile of f16 (x contiguous, y stride
// strideY elems, z stride strideZ elems) into LDS with 16B pad per 64B row.
__device__ __forceinline__ void tdm_tile_load(
    const _Float16* gptr, unsigned ldsOff,
    unsigned tileX, unsigned tileY, unsigned tileZ,
    unsigned long long strideY, unsigned long long strideZ,
    unsigned tensorX, unsigned tensorY, unsigned tensorZ) {
  unsigned long long ga = (unsigned long long)(uintptr_t)gptr;
  u32x4 g0 = {
      1u,                                                  // count=1, user D#
      ldsOff,                                              // lds_addr
      (unsigned)(ga & 0xffffffffu),                        // global_addr[31:0]
      (unsigned)((ga >> 32) & 0x01ffffffu) | (2u << 30)    // ga[56:32]|type=2
  };
  i32x8 g1 = {
      (int)((1u << 16)     // data_size = 2 bytes
            | (1u << 20)   // pad_enable
            | (3u << 22)   // pad_interval: 16 DWORDs (64B)
            | (3u << 25)), // pad_amount: 4 DWORDs (16B)
      (int)((tensorX & 0xffffu) << 16),                    // tensor_dim0 lo
      (int)((tensorX >> 16) | ((tensorY & 0xffffu) << 16)),// dim0 hi | dim1 lo
      (int)((tensorY >> 16) | (tileX << 16)),              // dim1 hi | tile0
      (int)(tileY | (tileZ << 16)),                        // tile1 | tile2
      (int)(strideY & 0xffffffffu),                        // dim0_stride lo
      (int)(((strideY >> 32) & 0xffffu) |
            ((strideZ & 0xffffu) << 16)),                  // d0s hi | d1s lo
      (int)((strideZ >> 16) & 0xffffffffu)                 // dim1_stride hi
  };
  i32x4 g2 = { (int)tensorZ, 0, 0, 0 };
  i32x4 g3 = { 0, 0, 0, 0 };
  i32x8 g4 = { 0, 0, 0, 0, 0, 0, 0, 0 };   // unused on this arity
  __builtin_amdgcn_tensor_load_to_lds(g0, g1, g2, g3, g4, 0);
}
#endif

template <int N>
__device__ __forceinline__ void wait_tdm() {
#if USE_TDM
  __builtin_amdgcn_s_wait_tensorcnt(N);
#endif
}

// ---------------------------------------------------------------------------
// f32 -> f16 conversion, 8 elements per thread
// ---------------------------------------------------------------------------
__global__ void cvt_f32_f16(const float* __restrict__ src,
                            _Float16* __restrict__ dst, int n8) {
  int i = blockIdx.x * blockDim.x + threadIdx.x;
  if (i >= n8) return;
  const float4* s = (const float4*)src + (size_t)i * 2;
  float4 a = s[0], b = s[1];
  v8h o;
  o[0] = (_Float16)a.x; o[1] = (_Float16)a.y;
  o[2] = (_Float16)a.z; o[3] = (_Float16)a.w;
  o[4] = (_Float16)b.x; o[5] = (_Float16)b.y;
  o[6] = (_Float16)b.z; o[7] = (_Float16)b.w;
  ((v8h*)dst)[i] = o;
}

// ---------------------------------------------------------------------------
// Gating GEMM: h = relu(x @ W_g1^T + b_g1)   [N_TOK, HDIM]
// 128x128 tile, BK=32, 8 waves of 32x64; TDM double-buffered staging.
// ---------------------------------------------------------------------------
__device__ __forceinline__ void gate_stage(
    int kst, const _Float16* __restrict__ xh, const _Float16* __restrict__ w1h,
    int nBase, int oBase, int tid, int waveU,
    _Float16 (&As)[2][128 * LDS_STRIDE],
    _Float16 (&Bs)[2][128 * LDS_STRIDE]) {
  const int k0 = kst * 32;
  const int buf = kst & 1;
#if USE_TDM
  if (waveU == 0) {
    tdm_tile_load(&xh[(size_t)nBase * DDIM + k0], lds_off(&As[buf][0]),
                  /*tile*/ 32, 128, 0,
                  /*strides*/ (unsigned long long)DDIM, 0ull,
                  /*tensor*/ DDIM, N_TOK, 0);
    tdm_tile_load(&w1h[(size_t)oBase * DDIM + k0], lds_off(&Bs[buf][0]),
                  /*tile*/ 32, 128, 0,
                  /*strides*/ (unsigned long long)DDIM, 0ull,
                  /*tensor*/ DDIM, HDIM, 0);
  }
#else
  (void)waveU;
#pragma unroll
  for (int j = 0; j < 2; ++j) {
    int c = tid * 2 + j;
    int row = c >> 2, ch = (c & 3) * 8;
    *(v8h*)&As[buf][row * LDS_STRIDE + ch] =
        *(const v8h*)&xh[(size_t)(nBase + row) * DDIM + k0 + ch];
    *(v8h*)&Bs[buf][row * LDS_STRIDE + ch] =
        *(const v8h*)&w1h[(size_t)(oBase + row) * DDIM + k0 + ch];
  }
#endif
}

__global__ __launch_bounds__(256)
void gate_h_gemm(const _Float16* __restrict__ xh,
                 const _Float16* __restrict__ w1h,
                 const float* __restrict__ bg1,
                 float* __restrict__ hout) {
  __shared__ __align__(16) _Float16 As[2][128 * LDS_STRIDE];   // 20 KB
  __shared__ __align__(16) _Float16 Bs[2][128 * LDS_STRIDE];   // 20 KB

  const int tid   = threadIdx.x;
  const int lane  = tid & 31;
  const int waveU = __builtin_amdgcn_readfirstlane(tid >> 5);
  const int wave  = tid >> 5;
  const int waveM = (wave & 3) * 32;
  const int waveN = (wave >> 2) * 64;
  const int nBase = blockIdx.y * 128;
  const int oBase = blockIdx.x * 128;
  const int ks    = ((lane >> 4) & 1) * 8;
  const int frow  = lane & 15;

  v8f acc[2][4] = {};

  gate_stage(0, xh, w1h, nBase, oBase, tid, waveU, As, Bs);

  for (int kst = 0; kst < 32; ++kst) {
    if (kst + 1 < 32) {
      gate_stage(kst + 1, xh, w1h, nBase, oBase, tid, waveU, As, Bs);
      wait_tdm<2>();            // allow the 2 just-issued descriptors in flight
    } else {
      wait_tdm<0>();
    }
    __syncthreads();

    const int buf = kst & 1;
    HF16 a[2], b[4];
#pragma unroll
    for (int mt = 0; mt < 2; ++mt) {
      int r = waveM + mt * 16 + frow;
      a[mt].h[0] = *(const v8h*)&As[buf][r * LDS_STRIDE + ks];
      a[mt].h[1] = *(const v8h*)&As[buf][r * LDS_STRIDE + ks + 16];
    }
#pragma unroll
    for (int nt = 0; nt < 4; ++nt) {
      int r = waveN + nt * 16 + frow;
      b[nt].h[0] = *(const v8h*)&Bs[buf][r * LDS_STRIDE + ks];
      b[nt].h[1] = *(const v8h*)&Bs[buf][r * LDS_STRIDE + ks + 16];
    }
#pragma unroll
    for (int nt = 0; nt < 4; ++nt)
#pragma unroll
      for (int mt = 0; mt < 2; ++mt)
        acc[mt][nt] = __builtin_amdgcn_wmma_f32_16x16x32_f16(
            false, a[mt].v, false, b[nt].v, (short)0, acc[mt][nt], false, false);
    __syncthreads();
  }

#pragma unroll
  for (int mt = 0; mt < 2; ++mt)
#pragma unroll
    for (int nt = 0; nt < 4; ++nt) {
      int col  = oBase + waveN + nt * 16 + frow;
      float bb = bg1[col];
      int row0 = nBase + waveM + mt * 16 + ((lane >> 4) << 3);
#pragma unroll
      for (int v = 0; v < 8; ++v) {
        float r = acc[mt][nt][v] + bb;
        hout[(size_t)(row0 + v) * HDIM + col] = r > 0.0f ? r : 0.0f;
      }
    }
}

// ---------------------------------------------------------------------------
// gates = softmax(h @ W_g2^T + b_g2); one thread per row
// ---------------------------------------------------------------------------
__global__ __launch_bounds__(256)
void gate_softmax(const float* __restrict__ h, const float* __restrict__ w2,
                  const float* __restrict__ b2, float* __restrict__ gates) {
  __shared__ float W2s[NEXP * HDIM];
  __shared__ float b2s[NEXP];
  int tid = threadIdx.x;
  for (int i = tid; i < NEXP * HDIM; i += 256) W2s[i] = w2[i];
  if (tid < NEXP) b2s[tid] = b2[tid];
  __syncthreads();

  int n = blockIdx.x * 256 + tid;
  float logit[NEXP];
#pragma unroll
  for (int e = 0; e < NEXP; ++e) logit[e] = b2s[e];
  const float* hr = h + (size_t)n * HDIM;
  for (int k = 0; k < HDIM; ++k) {
    float xv = hr[k];
#pragma unroll
    for (int e = 0; e < NEXP; ++e) logit[e] += xv * W2s[e * HDIM + k];
  }
  float mx = logit[0];
#pragma unroll
  for (int e = 1; e < NEXP; ++e) mx = fmaxf(mx, logit[e]);
  float p[NEXP], s = 0.0f;
#pragma unroll
  for (int e = 0; e < NEXP; ++e) { p[e] = __expf(logit[e] - mx); s += p[e]; }
  float inv = 1.0f / s;
#pragma unroll
  for (int e = 0; e < NEXP; ++e) gates[(size_t)n * NEXP + e] = p[e] * inv;
}

// ---------------------------------------------------------------------------
// out[n,o] = sum_e gates[n,e] * b_e[e,o]   (initializes d_out)
// ---------------------------------------------------------------------------
__global__ __launch_bounds__(256)
void bias_init(const float* __restrict__ gates, const float* __restrict__ be,
               float* __restrict__ out) {
  int n = blockIdx.x;
  int o = threadIdx.x * 4;
  float g[NEXP];
#pragma unroll
  for (int e = 0; e < NEXP; ++e) g[e] = gates[(size_t)n * NEXP + e];
  float4 r = make_float4(0.f, 0.f, 0.f, 0.f);
#pragma unroll
  for (int e = 0; e < NEXP; ++e) {
    float4 bv = *(const float4*)&be[(size_t)e * ODIM + o];
    r.x += g[e] * bv.x; r.y += g[e] * bv.y;
    r.z += g[e] * bv.z; r.w += g[e] * bv.w;
  }
  *(float4*)&out[(size_t)n * ODIM + o] = r;
}

// ---------------------------------------------------------------------------
// Main fused MoE GEMM helpers
// ---------------------------------------------------------------------------
// Stage one pipeline step: A tile (ph==0 only) + 4-expert B slab.
// TDM path: wave 0 issues descriptor-driven DMA. Fallback: per-thread copies.
__device__ __forceinline__ void stage_issue(
    int ph, int kst, const _Float16* __restrict__ xh,
    const _Float16* __restrict__ weh, int nBase, int oBase, int tid, int waveU,
    _Float16 (&As)[2][128 * LDS_STRIDE],
    _Float16 (&Bs)[2][4][128 * LDS_STRIDE]) {
  const int k0 = kst * 32;
#if USE_TDM
  if (waveU == 0) {
    if (ph == 0) {
      tdm_tile_load(&xh[(size_t)nBase * DDIM + k0], lds_off(&As[kst & 1][0]),
                    /*tile*/ 32, 128, 0,
                    /*strides*/ (unsigned long long)DDIM, 0ull,
                    /*tensor*/ DDIM, N_TOK, 0);
    }
    tdm_tile_load(&weh[(size_t)(ph * 4 * ODIM + oBase) * DDIM + k0],
                  lds_off(&Bs[ph][0][0]),
                  /*tile*/ 32, 128, 4,
                  /*strides*/ (unsigned long long)DDIM,
                  (unsigned long long)ODIM * DDIM,
                  /*tensor*/ DDIM, NEXP * ODIM, NEXP);
  }
#else
  (void)waveU;
  if (ph == 0) {
    const int ab = kst & 1;
#pragma unroll
    for (int j = 0; j < 2; ++j) {
      int c = tid * 2 + j;
      int row = c >> 2, ch = (c & 3) * 8;
      *(v8h*)&As[ab][row * LDS_STRIDE + ch] =
          *(const v8h*)&xh[(size_t)(nBase + row) * DDIM + k0 + ch];
    }
  }
#pragma unroll
  for (int i = 0; i < 8; ++i) {
    int c = tid + 256 * i;
    int eg = c >> 9;
    int cc = c & 511;
    int row = cc >> 2, ch = (cc & 3) * 8;
    int e = ph * 4 + eg;
    *(v8h*)&Bs[ph][eg][row * LDS_STRIDE + ch] =
        *(const v8h*)&weh[(size_t)(e * ODIM + oBase + row) * DDIM + k0 + ch];
  }
#endif
}

// 4 experts of gate-scaled WMMA (PH selects expert group at compile time so
// the gate register array is statically indexed).
template <int PH>
__device__ __forceinline__ void compute4(
    const HF16 (&a)[2], const _Float16 (&Bslab)[4][128 * LDS_STRIDE],
    const _Float16 (&gh)[2][NEXP], v8f (&acc)[2][4],
    int waveN, int frow, int ks) {
#pragma unroll
  for (int eg = 0; eg < 4; ++eg) {
    const int e = PH * 4 + eg;
    HF16 b[4];
#pragma unroll
    for (int nt = 0; nt < 4; ++nt) {
      int r = waveN + nt * 16 + frow;
      b[nt].h[0] = *(const v8h*)&Bslab[eg][r * LDS_STRIDE + ks];
      b[nt].h[1] = *(const v8h*)&Bslab[eg][r * LDS_STRIDE + ks + 16];
    }
    v16h as0 = a[0].v * gh[0][e];   // v_pk_mul_f16 gate scaling
    v16h as1 = a[1].v * gh[1][e];
#pragma unroll
    for (int nt = 0; nt < 4; ++nt) {
      acc[0][nt] = __builtin_amdgcn_wmma_f32_16x16x32_f16(
          false, as0, false, b[nt].v, (short)0, acc[0][nt], false, false);
      acc[1][nt] = __builtin_amdgcn_wmma_f32_16x16x32_f16(
          false, as1, false, b[nt].v, (short)0, acc[1][nt], false, false);
    }
  }
}

// ---------------------------------------------------------------------------
// Fused MoE GEMM: out += sum_e (g_e * x) @ W_e^T
// 128x128 tile, BK=32, 8 waves of 32x64. Double-buffered A (per K-step) and
// 4-expert B slab (per phase); TDM prefetch of stage s+1 overlaps the 32
// WMMAs of stage s, synchronized with s_wait_tensorcnt.
// ---------------------------------------------------------------------------
__global__ __launch_bounds__(256)
void moe_gemm(const _Float16* __restrict__ xh,
              const _Float16* __restrict__ weh,
              const float* __restrict__ gates,
              float* __restrict__ out) {
  __shared__ __align__(16) _Float16 As[2][128 * LDS_STRIDE];       // 20 KB
  __shared__ __align__(16) _Float16 Bs[2][4][128 * LDS_STRIDE];    // 80 KB

  const int tid   = threadIdx.x;
  const int lane  = tid & 31;
  const int waveU = __builtin_amdgcn_readfirstlane(tid >> 5);  // scalar branch
  const int wave  = tid >> 5;
  const int waveM = (wave & 3) * 32;
  const int waveN = (wave >> 2) * 64;
  const int nBase = blockIdx.y * 128;
  const int oBase = blockIdx.x * 128;
  const int ks    = ((lane >> 4) & 1) * 8;
  const int frow  = lane & 15;

  // Preload gates for the two A-fragment rows this lane owns, as f16.
  _Float16 gh[2][NEXP];
  {
    int r0 = nBase + waveM + frow;
#pragma unroll
    for (int e = 0; e < NEXP; ++e) {
      gh[0][e] = (_Float16)gates[(size_t)r0 * NEXP + e];
      gh[1][e] = (_Float16)gates[(size_t)(r0 + 16) * NEXP + e];
    }
  }

  v8f acc[2][4] = {};
  HF16 a[2];

  stage_issue(0, 0, xh, weh, nBase, oBase, tid, waveU, As, Bs);

  for (int kst = 0; kst < 32; ++kst) {
    // Prefetch phase-1 B slab, then compute phase 0 (experts 0..3).
    stage_issue(1, kst, xh, weh, nBase, oBase, tid, waveU, As, Bs);
    wait_tdm<1>();              // allow the 1 just-issued descriptor in flight
    __syncthreads();
    {
      const int ab = kst & 1;
#pragma unroll
      for (int mt = 0; mt < 2; ++mt) {
        int r = waveM + mt * 16 + frow;
        a[mt].h[0] = *(const v8h*)&As[ab][r * LDS_STRIDE + ks];
        a[mt].h[1] = *(const v8h*)&As[ab][r * LDS_STRIDE + ks + 16];
      }
      compute4<0>(a, Bs[0], gh, acc, waveN, frow, ks);
    }
    __syncthreads();

    // Prefetch next K-step (A + phase-0 B), then compute phase 1 (4..7).
    if (kst + 1 < 32) {
      stage_issue(0, kst + 1, xh, weh, nBase, oBase, tid, waveU, As, Bs);
      wait_tdm<2>();            // allow the 2 just-issued descriptors in flight
    } else {
      wait_tdm<0>();
    }
    __syncthreads();
    compute4<1>(a, Bs[1], gh, acc, waveN, frow, ks);
    __syncthreads();
  }

  // Epilogue: accumulate onto bias-initialized d_out.
#pragma unroll
  for (int mt = 0; mt < 2; ++mt)
#pragma unroll
    for (int nt = 0; nt < 4; ++nt) {
      int col  = oBase + waveN + nt * 16 + frow;
      int row0 = nBase + waveM + mt * 16 + ((lane >> 4) << 3);
#pragma unroll
      for (int v = 0; v < 8; ++v) {
        size_t idx = (size_t)(row0 + v) * ODIM + col;
        out[idx] += acc[mt][nt][v];
      }
    }
}

// ---------------------------------------------------------------------------
extern "C" void kernel_launch(void* const* d_in, const int* in_sizes, int n_in,
                              void* d_out, int out_size, void* d_ws, size_t ws_size,
                              hipStream_t stream) {
  (void)in_sizes; (void)n_in; (void)out_size; (void)ws_size;

  const float* x   = (const float*)d_in[0];  // [N, D]
  const float* We  = (const float*)d_in[1];  // [E, O, D]
  const float* be  = (const float*)d_in[2];  // [E, O]
  const float* Wg1 = (const float*)d_in[3];  // [H, D]
  const float* bg1 = (const float*)d_in[4];  // [H]
  const float* Wg2 = (const float*)d_in[5];  // [E, H]
  const float* bg2 = (const float*)d_in[6];  // [E]
  float* out = (float*)d_out;                // [N, O]

  // Workspace partition (~41 MB total)
  char* w = (char*)d_ws;
  _Float16* xh  = (_Float16*)w; w += (size_t)N_TOK * DDIM * 2;
  _Float16* weh = (_Float16*)w; w += (size_t)NEXP * ODIM * DDIM * 2;
  _Float16* w1h = (_Float16*)w; w += (size_t)HDIM * DDIM * 2;
  float* hbuf   = (float*)w;    w += (size_t)N_TOK * HDIM * 4;
  float* gbuf   = (float*)w;    w += (size_t)N_TOK * NEXP * 4;

  // 1) precision conversion
  cvt_f32_f16<<<(N_TOK * DDIM) / 2048, 256, 0, stream>>>(x, xh, (N_TOK * DDIM) / 8);
  cvt_f32_f16<<<(NEXP * ODIM * DDIM) / 2048, 256, 0, stream>>>(We, weh, (NEXP * ODIM * DDIM) / 8);
  cvt_f32_f16<<<(HDIM * DDIM) / 2048, 256, 0, stream>>>(Wg1, w1h, (HDIM * DDIM) / 8);

  // 2) gating network
  gate_h_gemm<<<dim3(HDIM / 128, N_TOK / 128), 256, 0, stream>>>(xh, w1h, bg1, hbuf);
  gate_softmax<<<N_TOK / 256, 256, 0, stream>>>(hbuf, Wg2, bg2, gbuf);

  // 3) gate-weighted expert bias -> d_out
  bias_init<<<N_TOK, 256, 0, stream>>>(gbuf, be, out);

  // 4) fused gate-weighted all-expert GEMM, accumulated onto d_out
  moe_gemm<<<dim3(ODIM / 128, N_TOK / 128), 256, 0, stream>>>(xh, weh, gbuf, out);
}
